// HumanVAttention_28784870818151
// MI455X (gfx1250) — matrix operand order repaired and checked by reading, not compiled, not run on hardware
//
#include <hip/hip_runtime.h>
#include <hip/hip_bf16.h>

#define USE_TDM 1   // Tensor Data Mover staging (set 0 to fall back to manual)

// ---------------------------------------------------------------------------
// Problem constants (from reference)
// ---------------------------------------------------------------------------
#define B_    2
#define S_    4096
#define H_    2048
#define NH_   16
#define NKV_  4
#define D_    128
#define BS_   64
#define QB_   64      // S_/BS_
#define SEL_  12      // GLOB + LOCAL
#define GLOB_ 4
#define LOCAL_ 8
#define STRIDE_ 16
#define MTOK_ (B_*S_)         // 8192 tokens
#define QDIM_ (NH_*D_)        // 2048
#define KVDIM_ (NKV_*D_)      // 512

static __device__ __forceinline__ float SCALEF() { return 0.08838834764831845f; } // D^-0.5
#define NEG_ (-1.0e9f)

typedef __bf16 bf16_t;
typedef __attribute__((ext_vector_type(16))) __bf16 v16bf;
typedef __attribute__((ext_vector_type(8)))  __bf16 v8bf;
typedef __attribute__((ext_vector_type(8)))  float  v8f;
typedef __attribute__((ext_vector_type(4)))  unsigned int u32x4;
typedef __attribute__((ext_vector_type(4)))  int i32x4;
typedef __attribute__((ext_vector_type(8)))  int i32x8;

static __device__ __forceinline__ bf16_t f2bf(float f) {
  unsigned u = __builtin_bit_cast(unsigned, f);
  unsigned short h = (unsigned short)((u + 0x7FFFu + ((u >> 16) & 1u)) >> 16);
  return __builtin_bit_cast(bf16_t, h);
}
static __device__ __forceinline__ unsigned short f2bfu(float f) {
  unsigned u = __builtin_bit_cast(unsigned, f);
  return (unsigned short)((u + 0x7FFFu + ((u >> 16) & 1u)) >> 16);
}
static __device__ __forceinline__ float bf2f(bf16_t b) {
  unsigned short h = __builtin_bit_cast(unsigned short, b);
  unsigned u = ((unsigned)h) << 16;
  return __builtin_bit_cast(float, u);
}
static __device__ __forceinline__ v16bf cat16(v8bf lo, v8bf hi) {
  return __builtin_shufflevector(lo, hi, 0,1,2,3,4,5,6,7,8,9,10,11,12,13,14,15);
}
static __device__ __forceinline__ v8f wmma_bf16(v16bf a, v16bf b, v8f c) {
  return __builtin_amdgcn_wmma_f32_16x16x32_bf16(false, a, false, b, (short)0, c,
                                                 false, false);
}

#if USE_TDM
// ---------------------------------------------------------------------------
// TDM: async DMA of a 2D tile of 2-byte elements (global, row-major, given row
// stride) into LDS (contiguous row-major). D# packing per ISA 08_async_tensor
// §8: group0 = {count=1, lds_addr, global_addr[56:0], type=2},
// group1 = {mask=0, data_size=1(2B), tensor_dim0/1, tile_dim0/1, dim0_stride}.
// Tracked with TENSORcnt. Issue from ONE wave only.
// ---------------------------------------------------------------------------
static __device__ __forceinline__ void tdm_load_2d(void* lds_dst, const void* gsrc,
                                                   unsigned width_elems,
                                                   unsigned height_rows,
                                                   unsigned row_stride_elems)
{
  unsigned long long ga = (unsigned long long)(size_t)gsrc;
  u32x4 g0;
  g0.x = 1u;                                            // count=1, user desc
  g0.y = (unsigned)(size_t)lds_dst;                     // LDS byte offset
  g0.z = (unsigned)(ga & 0xFFFFFFFFull);                // global_addr[31:0]
  g0.w = (unsigned)((ga >> 32) & 0x01FFFFFFull) | (2u << 30); // [56:32] | type=2
  i32x8 g1;
  g1[0] = (int)(1u << 16);                              // data_size = 2 bytes
  g1[1] = (int)((width_elems & 0xFFFFu) << 16);         // tensor_dim0 lo16
  g1[2] = (int)((width_elems >> 16) & 0xFFFFu)
        | (int)((height_rows & 0xFFFFu) << 16);         // dim0 hi | tensor_dim1 lo
  g1[3] = (int)((height_rows >> 16) & 0xFFFFu)
        | (int)((width_elems & 0xFFFFu) << 16);         // dim1 hi | tile_dim0
  g1[4] = (int)(height_rows & 0xFFFFu);                 // tile_dim1 (tile_dim2=0)
  g1[5] = (int)row_stride_elems;                        // tensor_dim0_stride lo32
  g1[6] = 0;
  g1[7] = 0;
  i32x4 z4 = (i32x4){0, 0, 0, 0};
#if defined(__clang_major__) && (__clang_major__ >= 23)
  i32x8 z8 = (i32x8){0, 0, 0, 0, 0, 0, 0, 0};
  __builtin_amdgcn_tensor_load_to_lds(g0, g1, z4, z4, z8, 0);
#else
  __builtin_amdgcn_tensor_load_to_lds(g0, g1, z4, z4, 0);
#endif
}
#endif

// ---------------------------------------------------------------------------
// fp32 -> bf16 conversion (vectorized, n must be a multiple of 4)
// ---------------------------------------------------------------------------
__global__ void f32_to_bf16_kernel(const float* __restrict__ src,
                                   bf16_t* __restrict__ dst, long n) {
  long stride = (long)gridDim.x * blockDim.x;
  for (long i = ((long)blockIdx.x * blockDim.x + threadIdx.x) * 4; i < n;
       i += stride * 4) {
    float4 v = *(const float4*)&src[i];
    uint2 o;
    o.x = (unsigned)f2bfu(v.x) | ((unsigned)f2bfu(v.y) << 16);
    o.y = (unsigned)f2bfu(v.z) | ((unsigned)f2bfu(v.w) << 16);
    *(uint2*)&dst[i] = o;
  }
}

// ---------------------------------------------------------------------------
// Tiled bf16 WMMA GEMM:  C[M,N] = A[M,K] * W[N,K]^T   (A,W bf16 row-major)
// Block tile 128x128, 256 threads (8 waves); each wave: 32 rows x 64 cols
// (2x4 WMMA tiles). Double-buffered LDS; TDM streams the next K-tile while
// all 8 waves run WMMA on the current one.
// ---------------------------------------------------------------------------
template<bool OUT_BF16>
__global__ __launch_bounds__(256)
void gemm_bf16_kernel(const bf16_t* __restrict__ A,
                      const bf16_t* __restrict__ W,
                      void* __restrict__ Cout,
                      int M, int N, int K)
{
  __shared__ __align__(16) bf16_t As[2][128 * 32];
  __shared__ __align__(16) bf16_t Ws[2][128 * 32];

  const int tid  = threadIdx.x;
  const int wave = tid >> 5;
  const int lane = tid & 31;
  const int m0 = blockIdx.y * 128;
  const int n0 = blockIdx.x * 128;
  const int wrow = wave & 3;     // 32-row strip
  const int wcol = wave >> 2;    // 64-col strip

  v8f acc[2][4];
#pragma unroll
  for (int r = 0; r < 2; ++r)
#pragma unroll
    for (int c = 0; c < 4; ++c) acc[r][c] = (v8f){};

  const int arow0 = (wrow * 2 + 0) * 16 + (lane & 15);
  const int arow1 = (wrow * 2 + 1) * 16 + (lane & 15);
  const int akb   = (lane >> 4) * 16;
  const int wkb   = (lane >> 4) * 16;
  int ncol[4];
#pragma unroll
  for (int ct = 0; ct < 4; ++ct) ncol[ct] = (wcol * 4 + ct) * 16 + (lane & 15);

  const int KT = K >> 5;

#if USE_TDM
  if (wave == 0) {
    tdm_load_2d(&As[0][0], &A[(long)m0 * K], 32, 128, (unsigned)K);
    tdm_load_2d(&Ws[0][0], &W[(long)n0 * K], 32, 128, (unsigned)K);
    __builtin_amdgcn_s_wait_tensorcnt(0);
  }
  __syncthreads();

  for (int kt = 0; kt < KT; ++kt) {
    const int cur = kt & 1;
    const bool more = (kt + 1 < KT);
    if (more && wave == 0) {   // DMA next tile while everyone computes
      const long k0 = (long)(kt + 1) << 5;
      tdm_load_2d(&As[cur ^ 1][0], &A[(long)m0 * K + k0], 32, 128, (unsigned)K);
      tdm_load_2d(&Ws[cur ^ 1][0], &W[(long)n0 * K + k0], 32, 128, (unsigned)K);
    }

    v16bf af0 = cat16(*(const v8bf*)&As[cur][arow0 * 32 + akb],
                      *(const v8bf*)&As[cur][arow0 * 32 + akb + 8]);
    v16bf af1 = cat16(*(const v8bf*)&As[cur][arow1 * 32 + akb],
                      *(const v8bf*)&As[cur][arow1 * 32 + akb + 8]);
#pragma unroll
    for (int ct = 0; ct < 4; ++ct) {
      v16bf bf = *(const v16bf*)&Ws[cur][ncol[ct] * 32 + wkb];
      acc[0][ct] = wmma_bf16(af0, bf, acc[0][ct]);
      acc[1][ct] = wmma_bf16(af1, bf, acc[1][ct]);
    }

    if (more && wave == 0) __builtin_amdgcn_s_wait_tensorcnt(0);
    __syncthreads();
  }
#else
  const int r0 = tid >> 2;
  const int c0 = (tid & 3) * 8;
  const long arowg0 = (long)(m0 + r0) * K + c0;
  const long arowg1 = (long)(m0 + r0 + 64) * K + c0;
  const long wrowg0 = (long)(n0 + r0) * K + c0;
  const long wrowg1 = (long)(n0 + r0 + 64) * K + c0;
  const int lds0 = r0 * 32 + c0;
  const int lds1 = (r0 + 64) * 32 + c0;
  {
    uint4 ra0 = *(const uint4*)&A[arowg0];
    uint4 ra1 = *(const uint4*)&A[arowg1];
    uint4 rw0 = *(const uint4*)&W[wrowg0];
    uint4 rw1 = *(const uint4*)&W[wrowg1];
    *(uint4*)&As[0][lds0] = ra0; *(uint4*)&As[0][lds1] = ra1;
    *(uint4*)&Ws[0][lds0] = rw0; *(uint4*)&Ws[0][lds1] = rw1;
  }
  __syncthreads();
  for (int kt = 0; kt < KT; ++kt) {
    const int cur = kt & 1;
    uint4 ra0, ra1, rw0, rw1;
    const bool more = (kt + 1 < KT);
    if (more) {
      const long k0 = (long)(kt + 1) << 5;
      ra0 = *(const uint4*)&A[arowg0 + k0];
      ra1 = *(const uint4*)&A[arowg1 + k0];
      rw0 = *(const uint4*)&W[wrowg0 + k0];
      rw1 = *(const uint4*)&W[wrowg1 + k0];
    }
    v16bf af0 = cat16(*(const v8bf*)&As[cur][arow0 * 32 + akb],
                      *(const v8bf*)&As[cur][arow0 * 32 + akb + 8]);
    v16bf af1 = cat16(*(const v8bf*)&As[cur][arow1 * 32 + akb],
                      *(const v8bf*)&As[cur][arow1 * 32 + akb + 8]);
#pragma unroll
    for (int ct = 0; ct < 4; ++ct) {
      v16bf bf = *(const v16bf*)&Ws[cur][ncol[ct] * 32 + wkb];
      acc[0][ct] = wmma_bf16(af0, bf, acc[0][ct]);
      acc[1][ct] = wmma_bf16(af1, bf, acc[1][ct]);
    }
    if (more) {
      const int nxt = cur ^ 1;
      *(uint4*)&As[nxt][lds0] = ra0; *(uint4*)&As[nxt][lds1] = ra1;
      *(uint4*)&Ws[nxt][lds0] = rw0; *(uint4*)&Ws[nxt][lds1] = rw1;
    }
    __syncthreads();
  }
#endif

  const int lrow8 = (lane & 16) ? 8 : 0;
#pragma unroll
  for (int rt = 0; rt < 2; ++rt) {
    const int rbase = m0 + (wrow * 2 + rt) * 16 + lrow8;
#pragma unroll
    for (int ct = 0; ct < 4; ++ct) {
      const int col = n0 + ncol[ct];
#pragma unroll
      for (int e = 0; e < 8; ++e) {
        long row = (long)(rbase + e);
        if constexpr (OUT_BF16)
          ((bf16_t*)Cout)[row * N + col] = f2bf(acc[rt][ct][e]);
        else
          ((float*)Cout)[row * N + col] = acc[rt][ct][e];
      }
    }
  }
}

// ---------------------------------------------------------------------------
// In-place RoPE on bf16 activations  X:[MTOK, heads*128], cos/sin:[MTOK,128]
// ---------------------------------------------------------------------------
__global__ void rope_kernel(bf16_t* __restrict__ X,
                            const float* __restrict__ cosb,
                            const float* __restrict__ sinb, int heads)
{
  long total = (long)MTOK_ * heads * (D_ / 2);
  long stride = (long)gridDim.x * blockDim.x;
  for (long i = (long)blockIdx.x * blockDim.x + threadIdx.x; i < total; i += stride) {
    int d = (int)(i & 63);
    long t = i >> 6;
    int hh = (int)(t % heads);
    long tok = t / heads;
    long base = tok * (long)heads * D_ + (long)hh * D_;
    float c0 = cosb[tok * D_ + d],      s0 = sinb[tok * D_ + d];
    float c1 = cosb[tok * D_ + d + 64], s1 = sinb[tok * D_ + d + 64];
    float x0 = bf2f(X[base + d]);
    float x1 = bf2f(X[base + d + 64]);
    X[base + d]      = f2bf(x0 * c0 - x1 * s0);
    X[base + d + 64] = f2bf(x1 * c1 + x0 * s1);
  }
}

// ---------------------------------------------------------------------------
// Closed-form block selection (matches reference dedup/causal logic)
// ---------------------------------------------------------------------------
static __device__ __forceinline__ void sel_of(int j, int qb, int& kb, bool& kp) {
  if (j < GLOB_) { kb = j * STRIDE_; kp = (kb <= qb); }
  else { kb = qb - (SEL_ - 1 - j); if (kb < 0) kb = 0; kp = ((kb & 15) != 0); }
}

// ---------------------------------------------------------------------------
// Fused block-sparse attention: one workgroup (8 waves) per (b, h, qblock).
// TDM streams Q and V tiles into LDS; the V landing buffer aliases the score
// buffer (their live ranges are disjoint within an iteration).
// ---------------------------------------------------------------------------
__global__ __launch_bounds__(256)
void attn_kernel(const bf16_t* __restrict__ Q,   // [MTOK, 2048]
                 const bf16_t* __restrict__ Kt,  // [MTOK, 512]
                 const bf16_t* __restrict__ V,   // [MTOK, 512]
                 bf16_t* __restrict__ O)         // [MTOK, 2048]
{
  __shared__ __align__(16) bf16_t Qs[64 * 128];
  __shared__ __align__(32) bf16_t VT[128 * 64];          // VT[d][tok]
  __shared__ __align__(16) unsigned char SsRaw[64 * 64 * 4]; // fp32 scores | V tile
  __shared__ __align__(16) bf16_t Ps[64 * 64];
  __shared__ __align__(32) float rowmax[64];
  __shared__ __align__(32) float rowsum[64];
  __shared__ __align__(32) float rowcorr[64];
  __shared__ __align__(16) float partred[4 * 64];

  float*  Ss   = (float*)SsRaw;
  bf16_t* Vrow = (bf16_t*)SsRaw;   // 64 tok x 128 d, row-major TDM landing zone

  const int tid  = threadIdx.x;
  const int wave = tid >> 5;
  const int lane = tid & 31;
  const int bid = blockIdx.x;
  const int qb  = bid & 63;
  const int h   = (bid >> 6) & 15;
  const int b   = bid >> 10;
  const int kvh = h >> 2;

  // --- Q tile -> LDS ---
  const long qrowbase = ((long)(b * S_ + qb * 64)) * QDIM_ + (long)h * D_;
#if USE_TDM
  if (wave == 0) {
    tdm_load_2d(&Qs[0], &Q[qrowbase], 128, 64, QDIM_);
    __builtin_amdgcn_s_wait_tensorcnt(0);
  }
#else
  for (int c = tid; c < 64 * 16; c += 256) {
    int row = c >> 4, off = (c & 15) * 8;
    *(uint4*)&Qs[row * 128 + off] =
        *(const uint4*)&Q[qrowbase + (long)row * QDIM_ + off];
  }
#endif
  if (tid < 64) { rowmax[tid] = -3.0e38f; rowsum[tid] = 0.0f; }

  const int rtile = wave & 3;
  const int chalf = wave >> 2;
  const int arowq = rtile * 16 + (lane & 15);
  const int akb   = (lane >> 4) * 16;
  const int wkb   = (lane >> 4) * 16;
  const int lrow8 = ((lane & 16) ? 8 : 0);
  v8f o0 = {}, o1 = {}, o2 = {}, o3 = {};

  __syncthreads();

  // --- hoist Q fragments (invariant over key blocks): 4 K-steps ---
  v16bf qf[4];
#pragma unroll
  for (int ks = 0; ks < 4; ++ks) {
    const int d0 = ks * 32;
    qf[ks] = cat16(*(const v8bf*)&Qs[arowq * 128 + d0 + akb],
                   *(const v8bf*)&Qs[arowq * 128 + d0 + akb + 8]);
  }

  const int col0 = (chalf * 2 + 0) * 16 + (lane & 15);
  const int col1 = (chalf * 2 + 1) * 16 + (lane & 15);

  for (int j = 0; j < SEL_; ++j) {
    int kb; bool kp;
    sel_of(j, qb, kb, kp);
    if (!kp) continue;             // uniform across workgroup

    const long krowbase = ((long)(b * S_ + kb * 64)) * KVDIM_ + (long)kvh * D_;

    // prefetch next kept block's K stream
    {
      int kbn; bool kpn;
      sel_of(j + 1 < SEL_ ? j + 1 : j, qb, kbn, kpn);
      if (kpn && j + 1 < SEL_) {
        const long nbase = ((long)(b * S_ + kbn * 64)) * KVDIM_ + (long)kvh * D_;
        const long prow = (long)(tid >> 2) * KVDIM_ + (tid & 3) * 32;
        __builtin_prefetch(&Kt[nbase + prow], 0, 1);
      }
    }

    // --- V tile -> LDS (row-major via TDM), then LDS transpose -> VT ---
#if USE_TDM
    if (wave == 0) {
      tdm_load_2d(&Vrow[0], &V[krowbase], 128, 64, KVDIM_);
      __builtin_amdgcn_s_wait_tensorcnt(0);
    }
    __syncthreads();
    for (int idx = tid; idx < 64 * 128; idx += 256) {
      int tok = idx >> 7, d = idx & 127;
      VT[d * 64 + tok] = Vrow[idx];
    }
#else
    for (int idx = tid; idx < 64 * 128; idx += 256) {
      int tok = idx >> 7, d = idx & 127;
      VT[d * 64 + tok] = V[krowbase + (long)tok * KVDIM_ + d];
    }
#endif
    __syncthreads();

    // --- scores: S = Q * K^T  (64x64) ---
    v8f s0 = {}, s1 = {};
#pragma unroll
    for (int ks = 0; ks < 4; ++ks) {
      const int d0 = ks * 32;
      v16bf kb0 = *(const v16bf*)&Kt[krowbase + (long)col0 * KVDIM_ + d0 + wkb];
      v16bf kb1 = *(const v16bf*)&Kt[krowbase + (long)col1 * KVDIM_ + d0 + wkb];
      s0 = wmma_bf16(qf[ks], kb0, s0);
      s1 = wmma_bf16(qf[ks], kb1, s1);
    }
#pragma unroll
    for (int e = 0; e < 8; ++e) {
      int row = rtile * 16 + lrow8 + e;
      float v0 = s0[e] * SCALEF();
      float v1 = s1[e] * SCALEF();
      if (kb == qb) {
        if (col0 > row) v0 = NEG_;
        if (col1 > row) v1 = NEG_;
      }
      Ss[row * 64 + col0] = v0;
      Ss[row * 64 + col1] = v1;
    }
    __syncthreads();

    // --- online softmax (4 threads per row) ---
    {
      const int prow = tid & 63, ppart = tid >> 6;
      const float* srow = &Ss[prow * 64 + ppart * 16];
      float m = -3.0e38f;
#pragma unroll
      for (int c = 0; c < 16; ++c) m = fmaxf(m, srow[c]);
      partred[ppart * 64 + prow] = m;
    }
    __syncthreads();
    if (tid < 64) {
      float mo = rowmax[tid];
      float mn = fmaxf(fmaxf(partred[tid], partred[64 + tid]),
                       fmaxf(partred[128 + tid], partred[192 + tid]));
      mn = fmaxf(mo, mn);
      rowcorr[tid] = __expf(mo - mn);
      rowmax[tid] = mn;
    }
    __syncthreads();
    {
      const int prow = tid & 63, ppart = tid >> 6;
      const float mn = rowmax[prow];
      const float* srow = &Ss[prow * 64 + ppart * 16];
      bf16_t* prowp = &Ps[prow * 64 + ppart * 16];
      float sum = 0.0f;
#pragma unroll
      for (int c = 0; c < 16; ++c) {
        float p = __expf(srow[c] - mn);
        prowp[c] = f2bf(p);
        sum += p;
      }
      partred[ppart * 64 + prow] = sum;
    }
    __syncthreads();
    if (tid < 64) {
      rowsum[tid] = rowsum[tid] * rowcorr[tid] + partred[tid] + partred[64 + tid] +
                    partred[128 + tid] + partred[192 + tid];
    }
    __syncthreads();

    // --- rescale O accumulators, then O += P * V ---
    v8f corrv = *(const v8f*)&rowcorr[rtile * 16 + lrow8];
    o0 *= corrv; o1 *= corrv; o2 *= corrv; o3 *= corrv;

    const int db = chalf * 64;
#pragma unroll
    for (int ks = 0; ks < 2; ++ks) {
      const int t0 = ks * 32;
      v16bf pf = cat16(*(const v8bf*)&Ps[arowq * 64 + t0 + akb],
                       *(const v8bf*)&Ps[arowq * 64 + t0 + akb + 8]);
      const int tokb = t0 + (lane >> 4) * 16;
      v16bf vb0 = *(const v16bf*)&VT[(db + 0 * 16 + (lane & 15)) * 64 + tokb];
      v16bf vb1 = *(const v16bf*)&VT[(db + 1 * 16 + (lane & 15)) * 64 + tokb];
      v16bf vb2 = *(const v16bf*)&VT[(db + 2 * 16 + (lane & 15)) * 64 + tokb];
      v16bf vb3 = *(const v16bf*)&VT[(db + 3 * 16 + (lane & 15)) * 64 + tokb];
      o0 = wmma_bf16(pf, vb0, o0);
      o1 = wmma_bf16(pf, vb1, o1);
      o2 = wmma_bf16(pf, vb2, o2);
      o3 = wmma_bf16(pf, vb3, o3);
    }
    __syncthreads();
  }

  // --- finalize: divide by row sums, store bf16 ---
  v8f sumv = *(const v8f*)&rowsum[rtile * 16 + lrow8];
  const long orow0 = (long)(b * S_ + qb * 64);
  const int colb = h * D_ + chalf * 64 + (lane & 15);
#pragma unroll
  for (int e = 0; e < 8; ++e) {
    long r = orow0 + rtile * 16 + lrow8 + e;
    float inv = 1.0f / sumv[e];
    O[r * QDIM_ + colb + 0]  = f2bf(o0[e] * inv);
    O[r * QDIM_ + colb + 16] = f2bf(o1[e] * inv);
    O[r * QDIM_ + colb + 32] = f2bf(o2[e] * inv);
    O[r * QDIM_ + colb + 48] = f2bf(o3[e] * inv);
  }
}

// ---------------------------------------------------------------------------
// Host-side launcher
// ---------------------------------------------------------------------------
extern "C" void kernel_launch(void* const* d_in, const int* in_sizes, int n_in,
                              void* d_out, int out_size, void* d_ws, size_t ws_size,
                              hipStream_t stream) {
  (void)in_sizes; (void)n_in; (void)out_size; (void)ws_size;
  const float* hs   = (const float*)d_in[0];
  const float* cosb = (const float*)d_in[1];
  const float* sinb = (const float*)d_in[2];
  const float* Wq   = (const float*)d_in[3];
  const float* Wk   = (const float*)d_in[4];
  const float* Wv   = (const float*)d_in[5];
  const float* Wo   = (const float*)d_in[6];
  float* out = (float*)d_out;

  char* ws = (char*)d_ws;
  size_t off = 0;
  auto alloc = [&](size_t bytes) -> char* {
    char* p = ws + off;
    off += (bytes + 255) & ~(size_t)255;
    return p;
  };
  bf16_t* Xb  = (bf16_t*)alloc((size_t)MTOK_ * H_ * 2);
  bf16_t* Wqb = (bf16_t*)alloc((size_t)QDIM_ * H_ * 2);
  bf16_t* Wkb = (bf16_t*)alloc((size_t)KVDIM_ * H_ * 2);
  bf16_t* Wvb = (bf16_t*)alloc((size_t)KVDIM_ * H_ * 2);
  bf16_t* Wob = (bf16_t*)alloc((size_t)H_ * QDIM_ * 2);
  bf16_t* Qb  = (bf16_t*)alloc((size_t)MTOK_ * QDIM_ * 2);
  bf16_t* Kb  = (bf16_t*)alloc((size_t)MTOK_ * KVDIM_ * 2);
  bf16_t* Vb  = (bf16_t*)alloc((size_t)MTOK_ * KVDIM_ * 2);
  bf16_t* Ob  = (bf16_t*)alloc((size_t)MTOK_ * QDIM_ * 2);

  // 1. fp32 -> bf16 conversions
  f32_to_bf16_kernel<<<4096, 256, 0, stream>>>(hs, Xb, (long)MTOK_ * H_);
  f32_to_bf16_kernel<<<1024, 256, 0, stream>>>(Wq, Wqb, (long)QDIM_ * H_);
  f32_to_bf16_kernel<<<512,  256, 0, stream>>>(Wk, Wkb, (long)KVDIM_ * H_);
  f32_to_bf16_kernel<<<512,  256, 0, stream>>>(Wv, Wvb, (long)KVDIM_ * H_);
  f32_to_bf16_kernel<<<1024, 256, 0, stream>>>(Wo, Wob, (long)H_ * QDIM_);

  // 2. QKV projections (WMMA GEMM, bf16 out)
  gemm_bf16_kernel<true><<<dim3(QDIM_ / 128, MTOK_ / 128), 256, 0, stream>>>(
      Xb, Wqb, Qb, MTOK_, QDIM_, H_);
  gemm_bf16_kernel<true><<<dim3(KVDIM_ / 128, MTOK_ / 128), 256, 0, stream>>>(
      Xb, Wkb, Kb, MTOK_, KVDIM_, H_);
  gemm_bf16_kernel<true><<<dim3(KVDIM_ / 128, MTOK_ / 128), 256, 0, stream>>>(
      Xb, Wvb, Vb, MTOK_, KVDIM_, H_);

  // 3. RoPE (in place on Q, K)
  rope_kernel<<<4096, 256, 0, stream>>>(Qb, cosb, sinb, NH_);
  rope_kernel<<<1024, 256, 0, stream>>>(Kb, cosb, sinb, NKV_);

  // 4. Fused block-sparse attention
  attn_kernel<<<B_ * NH_ * QB_, 256, 0, stream>>>(Qb, Kb, Vb, Ob);

  // 5. Output projection (fp32 out)
  gemm_bf16_kernel<false><<<dim3(H_ / 128, MTOK_ / 128), 256, 0, stream>>>(
      Ob, Wob, out, MTOK_, H_, QDIM_);
}
